// MaskedCausalAttention_59811714564665
// MI455X (gfx1250) — matrix operand order repaired and checked by reading, not compiled, not run on hardware
//
#include <hip/hip_runtime.h>
#include <hip/hip_bf16.h>

typedef __bf16 v16bf __attribute__((ext_vector_type(16)));
typedef float  v8f   __attribute__((ext_vector_type(8)));

#define BB    4
#define TT    2048
#define HDIM  1024
#define NH    16
#define DH    64
#define MROWS (BB * TT)   // 8192 flattened (b,t) rows

// f32 -> bf16 round-to-nearest-even (bit trick, no scalar __bf16 arithmetic)
__device__ __forceinline__ unsigned short f2bf(float f) {
    union { float f; unsigned int u; } v; v.f = f;
    unsigned int r = v.u + 0x7FFFu + ((v.u >> 16) & 1u);
    return (unsigned short)(r >> 16);
}
__device__ __forceinline__ unsigned int pack2bf(float lo, float hi) {
    return (unsigned int)f2bf(lo) | ((unsigned int)f2bf(hi) << 16);
}

// A-fragment (16x32 bf16): per lane one contiguous 32B run (K = half*16 + e).
__device__ __forceinline__ v16bf lds_frag_a(const unsigned short* p0) {
    union { uint4 u[2]; v16bf v; } t;
    t.u[0] = *(const uint4*)(p0);
    t.u[1] = *(const uint4*)(p0 + 8);
    return t.v;
}
// B-fragment (32x16 bf16): per lane one contiguous 32B run (K = half*16 + e).
__device__ __forceinline__ v16bf lds_frag_b(const unsigned short* p0) {
    union { uint4 u[2]; v16bf v; } t;
    t.u[0] = *(const uint4*)(p0);
    t.u[1] = *(const uint4*)(p0 + 8);
    return t.v;
}

// ln(10000)/32
#define ROPE_LN 0.28782313662425576f

// ---------------------------------------------------------------------------
// Kernel 1: q/k/v projection (bf16 WMMA, f32 acc) + in-register RoPE epilogue.
// grid = (128 Mtiles, 16 heads, 3 {q,k,v}), block = 128 (4 waves).
// K-tile = 64 -> 8 WMMAs per wave per barrier pair, 16 iterations.
// q,k -> [B,N,T,D] bf16 ; v -> [B,N,D,T] bf16 (pre-transposed for P@V).
// ---------------------------------------------------------------------------
__global__ __launch_bounds__(128)
void qkv_rope_59811714564665(const float* __restrict__ x,
                             const float* __restrict__ Wq, const float* __restrict__ bq,
                             const float* __restrict__ Wk, const float* __restrict__ bk,
                             const float* __restrict__ Wv, const float* __restrict__ bv,
                             unsigned short* __restrict__ qb,
                             unsigned short* __restrict__ kb,
                             unsigned short* __restrict__ vb) {
    __shared__ __align__(16) unsigned short Xs[64 * 64];  // [m][k] bf16
    __shared__ __align__(16) unsigned short Ws[64 * 64];  // [n][k] bf16
    const int tid = threadIdx.x;
    const int w = tid >> 5, lane = tid & 31, nl = lane & 15, hl = lane >> 4;
    const int mt = blockIdx.x;
    const int h  = blockIdx.y;
    const int z  = blockIdx.z;
    const float* W    = (z == 0) ? Wq : (z == 1) ? Wk : Wv;
    const float* bias = (z == 0) ? bq : (z == 1) ? bk : bv;

    v8f acc[4] = {};
    for (int k0 = 0; k0 < HDIM; k0 += 64) {
        // fill 64x64 tiles: float2 load -> pack 2 bf16 -> one ds_store_b32
        #pragma unroll
        for (int i = 0; i < 16; ++i) {
            int p = i * 128 + tid;               // 0..2047 pairs
            int r = p >> 5, c2 = (p & 31) * 2;
            float2 xv = *(const float2*)&x[(size_t)(mt * 64 + r) * HDIM + k0 + c2];
            *(unsigned int*)&Xs[r * 64 + c2] = pack2bf(xv.x, xv.y);
            float2 wv = *(const float2*)&W[(size_t)(h * 64 + r) * HDIM + k0 + c2];
            *(unsigned int*)&Ws[r * 64 + c2] = pack2bf(wv.x, wv.y);
        }
        __syncthreads();
        #pragma unroll
        for (int dstep = 0; dstep < 2; ++dstep) {
            v16bf a = lds_frag_a(&Xs[(w * 16 + nl) * 64 + dstep * 32 + hl * 16]);
            #pragma unroll
            for (int ns = 0; ns < 4; ++ns) {
                v16bf bfr = lds_frag_b(&Ws[(ns * 16 + nl) * 64 + dstep * 32 + hl * 16]);
                acc[ns] = __builtin_amdgcn_wmma_f32_16x16x32_bf16(
                    false, a, false, bfr, (short)0, acc[ns], false, false);
            }
        }
        __syncthreads();
    }

    // Epilogue: bias + RoPE (partner column d^32 = acc[ns^2], same lane/vgpr).
    const int mbase = mt * 64 + w * 16;
    #pragma unroll
    for (int ns = 0; ns < 4; ++ns) {
        const int d     = ns * 16 + nl;
        const float bd    = bias[h * 64 + d];
        const float bpart = bias[h * 64 + (d ^ 32)];
        // inv_freq = 10000^(-(d&31)/32) = exp(-(d&31)*ln(10000)/32); one fast exp
        const float freq  = __expf(-(float)(d & 31) * ROPE_LN);
        #pragma unroll
        for (int r = 0; r < 8; ++r) {
            const int m = mbase + r + 8 * hl;
            const int b = m >> 11, t = m & (TT - 1);
            float val = acc[ns][r] + bd;
            if (z < 2) {                         // RoPE for q and k
                float partner = acc[ns ^ 2][r] + bpart;
                float sn, cs;
                __sincosf((float)t * freq, &sn, &cs);
                float rot = (d < 32) ? -partner : partner;
                val = val * cs + rot * sn;
                unsigned short* dst = (z == 0) ? qb : kb;
                dst[(((size_t)b * NH + h) * TT + t) * DH + d] = f2bf(val);
            } else {                             // v: store transposed [B,N,D,T]
                vb[(((size_t)b * NH + h) * DH + d) * TT + t] = f2bf(val);
            }
        }
    }
}

// ---------------------------------------------------------------------------
// Kernel 2: flash attention. grid = (32 Qtiles, 16 heads, 4 batch), block 128.
// S = Q*K^T (WMMA), online softmax (f32), O += P*V (WMMA). att -> [B,T,N,D].
// ---------------------------------------------------------------------------
__global__ __launch_bounds__(128)
void flash_attn_59811714564665(const unsigned short* __restrict__ qb,
                               const unsigned short* __restrict__ kb,
                               const unsigned short* __restrict__ vb,
                               unsigned short* __restrict__ ab) {
    __shared__ __align__(16) unsigned short Qs [64 * 64];     // [t][d]
    __shared__ __align__(16) unsigned short Ks [64 * 64];     // [s][d]
    __shared__ __align__(16) unsigned short Vts[64 * 64];     // [d][s]
    __shared__ __align__(16) unsigned short Ps [4 * 16 * 64]; // per-wave P staging
    const int tid = threadIdx.x;
    const int w = tid >> 5, lane = tid & 31, nl = lane & 15, hl = lane >> 4;
    const int qt = blockIdx.x, h = blockIdx.y, b = blockIdx.z;
    const int bh = b * NH + h;

    const unsigned short* qsrc = qb + ((size_t)bh * TT + qt * 64) * DH;  // contiguous 64x64 tile
    #pragma unroll
    for (int i = 0; i < 4; ++i) {
        int c = i * 128 + tid;                       // 512 uint4 chunks = 8KB
        ((uint4*)Qs)[c] = ((const uint4*)qsrc)[c];
    }
    __syncthreads();
    v16bf qa[2];
    #pragma unroll
    for (int dstep = 0; dstep < 2; ++dstep)
        qa[dstep] = lds_frag_a(&Qs[(w * 16 + nl) * 64 + dstep * 32 + hl * 16]);

    float mrow[8], lrow[8];
    #pragma unroll
    for (int r = 0; r < 8; ++r) { mrow[r] = -3.0e38f; lrow[r] = 0.0f; }
    v8f o[4] = {};
    const int tbase = qt * 64 + w * 16;

    for (int j = 0; j <= qt; ++j) {
        __syncthreads();
        const unsigned short* ksrc = kb + ((size_t)bh * TT + j * 64) * DH;
        #pragma unroll
        for (int i = 0; i < 4; ++i) {
            int c = i * 128 + tid;
            ((uint4*)Ks)[c] = ((const uint4*)ksrc)[c];
        }
        #pragma unroll
        for (int i = 0; i < 4; ++i) {
            int c = i * 128 + tid;                   // 8 chunks per 64-wide Vt row
            int d = c >> 3, off = (c & 7) * 8;
            *(uint4*)&Vts[d * 64 + off] =
                *(const uint4*)&vb[((size_t)bh * DH + d) * TT + j * 64 + off];
        }
        // prefetch next iteration's K/V tiles (global_prefetch_b8)
        if (j < qt) {
            __builtin_prefetch(ksrc + 64 * DH + tid * 32, 0, 3);
            __builtin_prefetch(vb + ((size_t)bh * DH + (tid >> 1)) * TT
                                  + (j + 1) * 64 + (tid & 1) * 32, 0, 3);
        }
        __syncthreads();

        // ---- S = Q * K^T ----
        v8f sv[4] = {};
        #pragma unroll
        for (int dstep = 0; dstep < 2; ++dstep) {
            #pragma unroll
            for (int ns = 0; ns < 4; ++ns) {
                v16bf bfr = lds_frag_b(&Ks[(ns * 16 + nl) * 64 + dstep * 32 + hl * 16]);
                sv[ns] = __builtin_amdgcn_wmma_f32_16x16x32_bf16(
                    false, qa[dstep], false, bfr, (short)0, sv[ns], false, false);
            }
        }
        // ---- scale + causal mask ----
        #pragma unroll
        for (int ns = 0; ns < 4; ++ns) {
            const int sg = j * 64 + ns * 16 + nl;
            #pragma unroll
            for (int r = 0; r < 8; ++r) {
                const int tg = tbase + r + 8 * hl;
                float v = sv[ns][r] * 0.125f;        // 1/sqrt(64)
                sv[ns][r] = (sg > tg) ? -3.0e38f : v;
            }
        }
        // ---- online softmax (each row lives in 16 lanes of one half-wave) ----
        #pragma unroll
        for (int r = 0; r < 8; ++r) {
            float rm = fmaxf(fmaxf(sv[0][r], sv[1][r]), fmaxf(sv[2][r], sv[3][r]));
            #pragma unroll
            for (int msk = 1; msk < 16; msk <<= 1) rm = fmaxf(rm, __shfl_xor(rm, msk, 32));
            float nm   = fmaxf(mrow[r], rm);
            float corr = __expf(mrow[r] - nm);
            mrow[r] = nm;
            lrow[r] *= corr;
            #pragma unroll
            for (int ns = 0; ns < 4; ++ns) o[ns][r] *= corr;
            float rs = 0.0f;
            #pragma unroll
            for (int ns = 0; ns < 4; ++ns) {
                float p = __expf(sv[ns][r] - nm);
                sv[ns][r] = p; rs += p;
            }
            #pragma unroll
            for (int msk = 1; msk < 16; msk <<= 1) rs += __shfl_xor(rs, msk, 32);
            lrow[r] += rs;
        }
        // ---- stage P (C-layout) -> per-wave LDS -> A-layout reload ----
        unsigned short* pw = &Ps[w * 16 * 64];
        #pragma unroll
        for (int ns = 0; ns < 4; ++ns)
            #pragma unroll
            for (int r = 0; r < 8; ++r)
                pw[(r + 8 * hl) * 64 + ns * 16 + nl] = f2bf(sv[ns][r]);
        asm volatile("s_wait_dscnt 0" ::: "memory");  // wave-local LDS RAW fence
        // ---- O += P * V ----
        #pragma unroll
        for (int ss = 0; ss < 2; ++ss) {
            v16bf pa = lds_frag_a(&pw[nl * 64 + ss * 32 + hl * 16]);
            #pragma unroll
            for (int ns = 0; ns < 4; ++ns) {
                v16bf bfr = lds_frag_b(&Vts[(ns * 16 + nl) * 64 + ss * 32 + hl * 16]);
                o[ns] = __builtin_amdgcn_wmma_f32_16x16x32_bf16(
                    false, pa, false, bfr, (short)0, o[ns], false, false);
            }
        }
    }
    // normalize, store att as [B,T,N,D] bf16 (ready for output projection GEMM)
    #pragma unroll
    for (int r = 0; r < 8; ++r) {
        const int t = tbase + r + 8 * hl;
        const float invl = 1.0f / lrow[r];
        #pragma unroll
        for (int ns = 0; ns < 4; ++ns) {
            const int d = ns * 16 + nl;
            ab[(((size_t)b * TT + t) * NH + h) * DH + d] = f2bf(o[ns][r] * invl);
        }
    }
}

// ---------------------------------------------------------------------------
// Kernel 3: output projection att[8192,1024](bf16) @ Wp^T + bp -> f32 out.
// grid = (128 Mtiles, 16 Ntiles), block 128. K-tile = 64.
// ---------------------------------------------------------------------------
__global__ __launch_bounds__(128)
void out_proj_59811714564665(const unsigned short* __restrict__ ab,
                             const float* __restrict__ Wp, const float* __restrict__ bp,
                             float* __restrict__ out) {
    __shared__ __align__(16) unsigned short Xs[64 * 64];
    __shared__ __align__(16) unsigned short Ws[64 * 64];
    const int tid = threadIdx.x;
    const int w = tid >> 5, lane = tid & 31, nl = lane & 15, hl = lane >> 4;
    const int mt = blockIdx.x, nt = blockIdx.y;

    v8f acc[4] = {};
    for (int k0 = 0; k0 < HDIM; k0 += 64) {
        #pragma unroll
        for (int i = 0; i < 16; ++i) {
            int p = i * 128 + tid;
            int r = p >> 5, c2 = (p & 31) * 2;
            // A already bf16: straight u32 copy
            *(unsigned int*)&Xs[r * 64 + c2] =
                *(const unsigned int*)&ab[(size_t)(mt * 64 + r) * HDIM + k0 + c2];
            float2 wv = *(const float2*)&Wp[(size_t)(nt * 64 + r) * HDIM + k0 + c2];
            *(unsigned int*)&Ws[r * 64 + c2] = pack2bf(wv.x, wv.y);
        }
        __syncthreads();
        #pragma unroll
        for (int dstep = 0; dstep < 2; ++dstep) {
            v16bf a = lds_frag_a(&Xs[(w * 16 + nl) * 64 + dstep * 32 + hl * 16]);
            #pragma unroll
            for (int ns = 0; ns < 4; ++ns) {
                v16bf bfr = lds_frag_b(&Ws[(ns * 16 + nl) * 64 + dstep * 32 + hl * 16]);
                acc[ns] = __builtin_amdgcn_wmma_f32_16x16x32_bf16(
                    false, a, false, bfr, (short)0, acc[ns], false, false);
            }
        }
        __syncthreads();
    }
    #pragma unroll
    for (int ns = 0; ns < 4; ++ns) {
        const int n = nt * 64 + ns * 16 + nl;
        const float bn = bp[n];
        #pragma unroll
        for (int r = 0; r < 8; ++r) {
            const int m = mt * 64 + w * 16 + r + 8 * hl;
            out[(size_t)m * HDIM + n] = acc[ns][r] + bn;
        }
    }
}

extern "C" void kernel_launch(void* const* d_in, const int* in_sizes, int n_in,
                              void* d_out, int out_size, void* d_ws, size_t ws_size,
                              hipStream_t stream) {
    const float* x  = (const float*)d_in[0];
    const float* Wq = (const float*)d_in[1];
    const float* bq = (const float*)d_in[2];
    const float* Wk = (const float*)d_in[3];
    const float* bk = (const float*)d_in[4];
    const float* Wv = (const float*)d_in[5];
    const float* bv = (const float*)d_in[6];
    const float* Wp = (const float*)d_in[7];
    const float* bp = (const float*)d_in[8];

    // Workspace: 4 bf16 buffers of 8192*1024 elements (16 MB each, 64 MB total)
    unsigned short* qb = (unsigned short*)d_ws;
    unsigned short* kb = qb + (size_t)MROWS * HDIM;
    unsigned short* vb = kb + (size_t)MROWS * HDIM;
    unsigned short* ab = vb + (size_t)MROWS * HDIM;

    qkv_rope_59811714564665<<<dim3(MROWS / 64, NH, 3), 128, 0, stream>>>(
        x, Wq, bq, Wk, bk, Wv, bv, qb, kb, vb);
    flash_attn_59811714564665<<<dim3(TT / 64, NH, BB), 128, 0, stream>>>(
        qb, kb, vb, ab);
    out_proj_59811714564665<<<dim3(MROWS / 64, HDIM / 64), 128, 0, stream>>>(
        ab, Wp, bp, (float*)d_out);
}
